// TransformerAggregator_76759655514592
// MI455X (gfx1250) — matrix operand order repaired and checked by reading, not compiled, not run on hardware
//
#include <hip/hip_runtime.h>
#include <math.h>

typedef __bf16 bf16;
typedef __bf16 v8bf  __attribute__((ext_vector_type(8)));
typedef __bf16 v16bf __attribute__((ext_vector_type(16)));
typedef float  v8f   __attribute__((ext_vector_type(8)));

#define NROOT 16384
#define KNBR  32
#define DOUT  100
#define NCT   7      // ceil(100/16) column tiles

// ---------------------------------------------------------------------------
// Split-bf16 pack of W[Ktot][100] into WMMA B-fragment layout (hi/lo planes).
// B 32x16 fragment: lane<16 -> col=ct*16+lane,    K = kt*32 + {0..7, 16..23}
//                   lane>=16 -> col=ct*16+lane-16, K = kt*32 + {8..15, 24..31}
// packed[((ct*KT32+kt)*32+lane)*16 + e];  col>=100 / k>=Ktot zero-padded.
// ---------------------------------------------------------------------------
__global__ void pack_w_bf16_kernel(const float* __restrict__ W,
                                   bf16* __restrict__ hi, bf16* __restrict__ lo,
                                   int KT32, int Ktot) {
  int idx = blockIdx.x * blockDim.x + threadIdx.x;
  int total = NCT * KT32 * 512;
  if (idx >= total) return;
  int e    = idx & 15;
  int lane = (idx >> 4) & 31;
  int t    = idx >> 9;          // ct*KT32 + kt
  int kt   = t % KT32;
  int ct   = t / KT32;
  int col  = ct * 16 + (lane & 15);
  int k    = kt * 32 + ((lane >> 4) << 3) + (e & 7) + ((e >> 3) << 4);
  float v = 0.f;
  if (col < DOUT && k < Ktot) v = W[(size_t)k * DOUT + col];
  bf16 h = (bf16)v;
  hi[idx] = h;
  lo[idx] = (bf16)(v - (float)h);
}

// ---------------------------------------------------------------------------
// C = concat(A0..A3) @ W + bias via split-bf16 WMMA 16x16x32:
//   A = Ah + Al, B = Bh + Bl;  C ~= Ah*Bh + Ah*Bl + Al*Bh  (fp32 accum, ~1e-5 rel)
// Each term gets its OWN accumulator -> consecutive WMMAs never share D
// (no XDL pipeline padding); partial sums added in the epilogue.
// Prefetch distance 2 on the fragment loads to cover L2 latency.
// Block = 224 threads = 7 waves, wave w owns column tile w.
// DUAL=true additionally computes a second output with the SAME A (fused K+V).
// ---------------------------------------------------------------------------
template <bool DUAL>
__global__ __launch_bounds__(224)
void gemm_bf16x3_kernel(const float* __restrict__ A0, int c0,
                        const float* __restrict__ A1, int c1,
                        const float* __restrict__ A2, int c2,
                        const float* __restrict__ A3, int c3,
                        int Ktot, int Kpad,
                        const bf16* __restrict__ BHa, const bf16* __restrict__ BLa,
                        const float* __restrict__ biasA, float* __restrict__ outA,
                        const bf16* __restrict__ BHb, const bf16* __restrict__ BLb,
                        const float* __restrict__ biasB, float* __restrict__ outB) {
  __shared__ bf16 sH[16 * 552];           // max strideE = 544+8
  __shared__ bf16 sL[16 * 552];
  const int strideE = Kpad + 8;           // mult of 8 -> 16B-aligned b128 rows,
                                          // bank starts spaced 4 -> conflict-free
  const int tid = threadIdx.x;

  // ---- stage 16 x Ktot A tile, converting fp32 -> bf16 hi/lo planes ----
  {
    const float* srcs[4] = {A0, A1, A2, A3};
    int          cols[4] = {c0, c1, c2, c3};
    int colbase = 0;
    for (int s = 0; s < 4; ++s) {
      int C = cols[s];
      if (C == 0) continue;
      const float* S = srcs[s] + (size_t)blockIdx.x * 16 * C;   // rows contiguous
      for (int idx = tid; idx < 16 * C; idx += 224) {
        int r = idx / C;
        int c = idx - r * C;
        float x = S[idx];                                       // coalesced
        bf16 h = (bf16)x;
        int o = r * strideE + colbase + c;
        sH[o] = h;
        sL[o] = (bf16)(x - (float)h);
      }
      colbase += C;
    }
    int pw = Kpad - Ktot;                 // zero-fill k padding
    if (pw > 0) {
      for (int idx = tid; idx < 16 * pw; idx += 224) {
        int r = idx / pw;
        int c = idx - r * pw;
        int o = r * strideE + Ktot + c;
        sH[o] = (bf16)0.f;
        sL[o] = (bf16)0.f;
      }
    }
  }
  __syncthreads();

  const int lane = tid & 31;
  const int wave = tid >> 5;              // == column tile (0..6)
  const int KT32 = Kpad >> 5;             // >= 9 for all our shapes

  // A fragment: lane<16 -> row=lane,   K kt*32 + {0..7, 16..23}
  //             lane>=16 -> row=l-16,  K kt*32 + {8..15, 24..31}
  const int abase = (lane & 15) * strideE + ((lane >> 4) << 3);
  const v16bf* bHa = (const v16bf*)BHa + (size_t)wave * KT32 * 32 + lane;
  const v16bf* bLa = (const v16bf*)BLa + (size_t)wave * KT32 * 32 + lane;
  const v16bf* bHb = DUAL ? ((const v16bf*)BHb + (size_t)wave * KT32 * 32 + lane) : bHa;
  const v16bf* bLb = DUAL ? ((const v16bf*)BLb + (size_t)wave * KT32 * 32 + lane) : bLa;

  // one accumulator per split term (and per output) -> independent WMMAs
  v8f a0A = {0.f,0.f,0.f,0.f,0.f,0.f,0.f,0.f};
  v8f a1A = a0A, a2A = a0A;
  v8f a0B = a0A, a1B = a0A, a2B = a0A;

  struct Frag { v16bf ah, al, wha, wla, whb, wlb; };

  auto load_chunk = [&](int kt, Frag& f) {
    int ix = abase + kt * 32;
    v8bf h0 = *(const v8bf*)&sH[ix];        // ds_load_b128
    v8bf h1 = *(const v8bf*)&sH[ix + 16];
    v8bf l0 = *(const v8bf*)&sL[ix];
    v8bf l1 = *(const v8bf*)&sL[ix + 16];
    f.ah = __builtin_shufflevector(h0, h1, 0,1,2,3,4,5,6,7,8,9,10,11,12,13,14,15);
    f.al = __builtin_shufflevector(l0, l1, 0,1,2,3,4,5,6,7,8,9,10,11,12,13,14,15);
    f.wha = bHa[kt * 32];                   // 32B contiguous, L2/WGP$-hot
    f.wla = bLa[kt * 32];
    if (DUAL) {
      f.whb = bHb[kt * 32];
      f.wlb = bLb[kt * 32];
    }
  };
  auto compute = [&](const Frag& f) {
    a0A = __builtin_amdgcn_wmma_f32_16x16x32_bf16(false, f.ah, false, f.wha, (short)0, a0A, false, false);
    if (DUAL)
      a0B = __builtin_amdgcn_wmma_f32_16x16x32_bf16(false, f.ah, false, f.whb, (short)0, a0B, false, false);
    a1A = __builtin_amdgcn_wmma_f32_16x16x32_bf16(false, f.ah, false, f.wla, (short)0, a1A, false, false);
    if (DUAL)
      a1B = __builtin_amdgcn_wmma_f32_16x16x32_bf16(false, f.ah, false, f.wlb, (short)0, a1B, false, false);
    a2A = __builtin_amdgcn_wmma_f32_16x16x32_bf16(false, f.al, false, f.wha, (short)0, a2A, false, false);
    if (DUAL)
      a2B = __builtin_amdgcn_wmma_f32_16x16x32_bf16(false, f.al, false, f.whb, (short)0, a2B, false, false);
  };

  // prefetch-distance-2 software pipeline
  Frag C, N, M;
  load_chunk(0, C);
  load_chunk(1, N);
  int kt = 0;
  #pragma unroll 1
  for (; kt + 2 < KT32; ++kt) {
    load_chunk(kt + 2, M);    // issue loads for kt+2 before computing kt
    compute(C);
    C = N;
    N = M;
  }
  compute(C);                 // peeled last two chunks (no in-loop guards)
  compute(N);

  // Epilogue: lane holds col = 16*ct + (lane&15); rows M = v + 8*(lane>=16).
  v8f accA = a0A + a1A + a2A;
  int col = wave * 16 + (lane & 15);
  if (col < DOUT) {
    int rbase = blockIdx.x * 16 + ((lane >> 4) << 3);
    float ba = biasA[col];
    #pragma unroll
    for (int v = 0; v < 8; ++v)
      outA[(size_t)(rbase + v) * DOUT + col] = accA[v] + ba;
    if (DUAL) {
      v8f accB = a0B + a1B + a2B;
      float bb = biasB[col];
      #pragma unroll
      for (int v = 0; v < 8; ++v)
        outB[(size_t)(rbase + v) * DOUT + col] = accB[v] + bb;
    }
  }
}

// ---------------------------------------------------------------------------
// Attention: block = 64 threads = 2 waves, one root per wave.
// Lane <-> neighbor for scores + shuffle softmax; lane <-> channel for V-sum.
// ---------------------------------------------------------------------------
__global__ __launch_bounds__(64)
void attn_kernel(const float* __restrict__ hq, const float* __restrict__ hk,
                 const float* __restrict__ hv, float* __restrict__ hneigh) {
  __shared__ float sk[2][KNBR * 101];
  __shared__ float sv[2][KNBR * 101];
  __shared__ float sq[2][DOUT];
  __shared__ float satt[2][KNBR * 2];
  const int tid = threadIdx.x;
  const int n0 = blockIdx.x * 2;
  const size_t base = (size_t)n0 * KNBR * DOUT;

  for (int idx = tid; idx < 2 * KNBR * DOUT; idx += 64) {
    int r = idx / DOUT;
    int c = idx - r * DOUT;
    int w = r >> 5, rr = r & 31;
    sk[w][rr * 101 + c] = hk[base + idx];
    sv[w][rr * 101 + c] = hv[base + idx];
  }
  for (int idx = tid; idx < 2 * DOUT; idx += 64)
    sq[idx / DOUT][idx % DOUT] = hq[(size_t)n0 * DOUT + idx];
  __syncthreads();

  const int w = tid >> 5;       // root within block
  const int k = tid & 31;       // neighbor
  float s0 = 0.f, s1 = 0.f;
  const float* kr = &sk[w][k * 101];
  #pragma unroll 10
  for (int j = 0; j < 50; ++j)  s0 += sq[w][j] * kr[j];
  #pragma unroll 10
  for (int j = 50; j < DOUT; ++j) s1 += sq[w][j] * kr[j];
  s0 = (s0 > 0.f) ? s0 : 0.2f * s0;               // leaky_relu(0.2)
  s1 = (s1 > 0.f) ? s1 : 0.2f * s1;

  float m0 = s0, m1 = s1;
  for (int off = 16; off > 0; off >>= 1) {
    m0 = fmaxf(m0, __shfl_xor(m0, off, 32));
    m1 = fmaxf(m1, __shfl_xor(m1, off, 32));
  }
  float e0 = expf(s0 - m0), e1 = expf(s1 - m1);
  float t0 = e0, t1 = e1;
  for (int off = 16; off > 0; off >>= 1) {
    t0 += __shfl_xor(t0, off, 32);
    t1 += __shfl_xor(t1, off, 32);
  }
  satt[w][k * 2 + 0] = e0 / t0;
  satt[w][k * 2 + 1] = e1 / t1;
  __syncthreads();

  for (int i = 0; i < 4; ++i) {
    int j = (tid & 31) + 32 * i;
    if (j < DOUT) {
      int h = (j >= 50);
      float acc = 0.f;
      #pragma unroll 8
      for (int kk = 0; kk < KNBR; ++kk)
        acc += satt[w][kk * 2 + h] * sv[w][kk * 101 + j];
      hneigh[(size_t)(n0 + w) * DOUT + j] = acc;
    }
  }
}

// ---------------------------------------------------------------------------
// ReLU + LayerNorm over 100 channels, one wave per row.
// ---------------------------------------------------------------------------
__global__ __launch_bounds__(256)
void relu_ln_kernel(const float* __restrict__ hpre, const float* __restrict__ g,
                    const float* __restrict__ b, float* __restrict__ out) {
  const int w = threadIdx.x >> 5;
  const int lane = threadIdx.x & 31;
  const int row = blockIdx.x * 8 + w;
  float x[4];
  float s = 0.f, s2 = 0.f;
  #pragma unroll
  for (int i = 0; i < 4; ++i) {
    int j = lane + 32 * i;
    float v = 0.f;
    if (j < DOUT) v = fmaxf(hpre[(size_t)row * DOUT + j], 0.f);
    x[i] = v;
    s += v;
    s2 += v * v;
  }
  for (int off = 16; off > 0; off >>= 1) {
    s  += __shfl_xor(s,  off, 32);
    s2 += __shfl_xor(s2, off, 32);
  }
  float mu  = s * (1.f / DOUT);
  float var = s2 * (1.f / DOUT) - mu * mu;
  float r = rsqrtf(var + 1e-5f);
  #pragma unroll
  for (int i = 0; i < 4; ++i) {
    int j = lane + 32 * i;
    if (j < DOUT)
      out[(size_t)row * DOUT + j] = (x[i] - mu) * r * g[j] + b[j];
  }
}

// ---------------------------------------------------------------------------
extern "C" void kernel_launch(void* const* d_in, const int* in_sizes, int n_in,
                              void* d_out, int out_size, void* d_ws, size_t ws_size,
                              hipStream_t stream) {
  const float* root_feat = (const float*)d_in[0];
  const float* nbr_feat  = (const float*)d_in[1];
  const float* nbr_edge  = (const float*)d_in[2];
  const float* zt        = (const float*)d_in[3];
  const float* et        = (const float*)d_in[4];
  const float* root_mem  = (const float*)d_in[5];
  const float* nbr_mem   = (const float*)d_in[6];
  const float* Wq = (const float*)d_in[7];
  const float* bq = (const float*)d_in[8];
  const float* Wk = (const float*)d_in[9];
  const float* bk = (const float*)d_in[10];
  const float* Wv = (const float*)d_in[11];
  const float* bv = (const float*)d_in[12];
  const float* Wo = (const float*)d_in[13];
  const float* bo = (const float*)d_in[14];
  const float* ln_g = (const float*)d_in[15];
  const float* ln_b = (const float*)d_in[16];
  float* out = (float*)d_out;

  // k-dims padded to multiples of 32
  const int KT32q = 12, KT32kv = 17, KT32o = 9;     // Kpad = 384 / 544 / 288

  char* wsb = (char*)d_ws;
  auto alloc = [&](size_t bytes) {
    void* p = (void*)wsb;
    wsb += (bytes + 255) & ~(size_t)255;
    return p;
  };
  float* hq     = (float*)alloc((size_t)NROOT * DOUT * 4);
  float* hk     = (float*)alloc((size_t)NROOT * KNBR * DOUT * 4);
  float* hv     = (float*)alloc((size_t)NROOT * KNBR * DOUT * 4);
  float* hneigh = (float*)alloc((size_t)NROOT * DOUT * 4);
  float* hpre   = (float*)alloc((size_t)NROOT * DOUT * 4);
  bf16* wqpH = (bf16*)alloc((size_t)NCT * KT32q  * 512 * 2);
  bf16* wqpL = (bf16*)alloc((size_t)NCT * KT32q  * 512 * 2);
  bf16* wkpH = (bf16*)alloc((size_t)NCT * KT32kv * 512 * 2);
  bf16* wkpL = (bf16*)alloc((size_t)NCT * KT32kv * 512 * 2);
  bf16* wvpH = (bf16*)alloc((size_t)NCT * KT32kv * 512 * 2);
  bf16* wvpL = (bf16*)alloc((size_t)NCT * KT32kv * 512 * 2);
  bf16* wopH = (bf16*)alloc((size_t)NCT * KT32o  * 512 * 2);
  bf16* wopL = (bf16*)alloc((size_t)NCT * KT32o  * 512 * 2);

  // 1) split-bf16 pack of the weights into WMMA B-fragment layout
  { int t = NCT * KT32q  * 512; pack_w_bf16_kernel<<<(t + 255) / 256, 256, 0, stream>>>(Wq, wqpH, wqpL, KT32q, 372); }
  { int t = NCT * KT32kv * 512; pack_w_bf16_kernel<<<(t + 255) / 256, 256, 0, stream>>>(Wk, wkpH, wkpL, KT32kv, 544); }
  { int t = NCT * KT32kv * 512; pack_w_bf16_kernel<<<(t + 255) / 256, 256, 0, stream>>>(Wv, wvpH, wvpL, KT32kv, 544); }
  { int t = NCT * KT32o  * 512; pack_w_bf16_kernel<<<(t + 255) / 256, 256, 0, stream>>>(Wo, wopH, wopL, KT32o, 272); }

  // 2) h_q = [root_feat | zero_time | root_mem] @ Wq + bq
  gemm_bf16x3_kernel<false><<<NROOT / 16, 224, 0, stream>>>(
      root_feat, 172, zt, 100, root_mem, 100, nullptr, 0, 372, KT32q * 32,
      wqpH, wqpL, bq, hq, nullptr, nullptr, nullptr, nullptr);

  // 3) fused h_k / h_v = [nbr_feat | nbr_edge | edge_time | nbr_mem] @ {Wk,Wv}
  //    (A staged/converted once -> halves the dominant 1.14 GB read)
  gemm_bf16x3_kernel<true><<<(NROOT * KNBR) / 16, 224, 0, stream>>>(
      nbr_feat, 172, nbr_edge, 172, et, 100, nbr_mem, 100, 544, KT32kv * 32,
      wkpH, wkpL, bk, hk, wvpH, wvpL, bv, hv);

  // 4) per-root 2-head attention over 32 neighbors
  attn_kernel<<<NROOT / 2, 64, 0, stream>>>(hq, hk, hv, hneigh);

  // 5) h_pre = [h_neigh | root_feat] @ Wo + bo
  gemm_bf16x3_kernel<false><<<NROOT / 16, 224, 0, stream>>>(
      hneigh, 100, root_feat, 172, nullptr, 0, nullptr, 0, 272, KT32o * 32,
      wopH, wopL, bo, hpre, nullptr, nullptr, nullptr, nullptr);

  // 6) ReLU + LayerNorm -> output
  relu_ln_kernel<<<NROOT / 8, 256, 0, stream>>>(hpre, ln_g, ln_b, out);
}
